// CustomLSTM_67422396613154
// MI455X (gfx1250) — matrix operand order repaired and checked by reading, not compiled, not run on hardware
//
#include <hip/hip_runtime.h>
#include <stdint.h>

// ---------------------------------------------------------------------------
// LSTM: x(1024,32,512) f32, W(512,2048), U(512,2048), bias(2048)
// out = [hidden_seq (1024*32*512) | h_T (32*512) | c_T (32*512)]  (f32)
//
// Plan:
//   K0: convert x -> bf16 (flat), W/U -> bf16 *transposed* (N-major, K contig)
//   K1: xW GEMM with v_wmma_f32_16x16x32_bf16, async global->LDS staging
//   K2: persistent 16-WG recurrence; U slice resident in LDS (CDNA5 320KB),
//       h exchanged via global bf16 double buffer + agent-scope grid barrier,
//       xW chunk async-prefetched into LDS overlapping the h@U WMMA loop.
// ---------------------------------------------------------------------------

typedef __attribute__((ext_vector_type(16))) __bf16 v16bf;
typedef __attribute__((ext_vector_type(8)))  float  v8f;

union FragB16 { int4 q2[2]; v16bf v; };

#define T_STEPS 1024
#define B_SZ    32
#define H_SZ    512
#define G4      2048   // 4*H

// workspace layout (bytes)
#define XW_OFF   ((size_t)0)                      // f32  1024*32*2048      = 256MB
#define XBF_OFF  ((size_t)268435456)              // bf16 1024*32*512       = 32MB
#define WBT_OFF  ((size_t)301989888)              // bf16 2048*512 (transposed)
#define UBT_OFF  ((size_t)304087040)              // bf16 2048*512 (transposed)
#define HBUF_OFF ((size_t)306184192)              // bf16 2 * 32*512
#define CTR_OFF  ((size_t)306249728)              // u32 barrier counter

__device__ __forceinline__ unsigned ldsAddr(const void* p) {
    // flat LDS pointer: low 32 bits are the LDS byte offset
    return (unsigned)(unsigned long long)p;
}

// async global -> LDS, 16 bytes per lane, GVS addressing (SGPR base + VGPR u32 offset)
__device__ __forceinline__ void async_g2l_b128(unsigned lds_byte_addr,
                                               const void* sbase,
                                               unsigned byte_off) {
    asm volatile("global_load_async_to_lds_b128 %0, %1, %2"
                 :: "v"(lds_byte_addr), "v"(byte_off), "s"(sbase)
                 : "memory");
}
__device__ __forceinline__ void wait_async_le4() {
    asm volatile("s_wait_asynccnt 0x4" ::: "memory");
}
__device__ __forceinline__ void wait_async_0() {
    asm volatile("s_wait_asynccnt 0x0" ::: "memory");
}

__device__ __forceinline__ float sigmoidf_(float x) {
    return 1.0f / (1.0f + __expf(-x));
}

// ---------------------------------------------------------------------------
// K0a: flat f32 -> bf16
// ---------------------------------------------------------------------------
__global__ void cvt_x_kernel(const float* __restrict__ in, __bf16* __restrict__ outb, int n) {
    int i = (blockIdx.x * blockDim.x + threadIdx.x) * 4;
    if (i + 3 < n) {
        float4 v = *(const float4*)(in + i);
        outb[i + 0] = (__bf16)v.x;
        outb[i + 1] = (__bf16)v.y;
        outb[i + 2] = (__bf16)v.z;
        outb[i + 3] = (__bf16)v.w;
    }
}

// ---------------------------------------------------------------------------
// K0b: (512,2048) f32 row-major -> (2048,512) bf16 row-major (transpose+convert)
// ---------------------------------------------------------------------------
__global__ void cvt_tr_kernel(const float* __restrict__ in, __bf16* __restrict__ outb) {
    int idx = blockIdx.x * blockDim.x + threadIdx.x;   // 0 .. 512*2048-1
    int k = idx >> 11;          // row in input (K)
    int n = idx & 2047;         // col in input (N)
    outb[n * 512 + k] = (__bf16)in[idx];
}

// ---------------------------------------------------------------------------
// K1: xW = Xbf(32768x512) @ WbT'(512x2048) + bias, f32 out.
//     BM=64 BN=256 BK=32; 8 waves; wave tile 32x64 (2x4 frags of 16x16).
// ---------------------------------------------------------------------------
__global__ __launch_bounds__(256) void xw_gemm_kernel(
    const __bf16* __restrict__ Xbf, const __bf16* __restrict__ WbT,
    const float* __restrict__ bias, float* __restrict__ XW) {

    __shared__ __bf16 Ab[2][64 * 32];   // 4KB  each
    __shared__ __bf16 Bb[2][256 * 32];  // 16KB each

    const int tid  = threadIdx.x;
    const int wave = tid >> 5;
    const int lane = tid & 31;
    const int hl   = lane >> 4;
    const int l15  = lane & 15;

    const int m0 = blockIdx.y * 64;
    const int n0 = blockIdx.x * 256;
    const int wm = (wave >> 2) * 32;   // 0 / 32
    const int wn = (wave & 3) * 64;    // 0..192

    v8f acc[2][4];
    for (int i = 0; i < 2; ++i)
        for (int j = 0; j < 4; ++j)
            acc[i][j] = (v8f)0.0f;

    auto stage = [&](int buf, int k0) {
        // A tile: 64 rows x 32 k, one 16B chunk per thread
        {
            int row = tid >> 2, ko = (tid & 3) * 8;
            async_g2l_b128(ldsAddr(&Ab[buf][row * 32 + ko]), Xbf,
                           (unsigned)(((m0 + row) * 512 + k0 + ko) * 2));
        }
        // B tile: 256 cols x 32 k, four 16B chunks per thread (from transposed W)
        for (int r = 0; r < 4; ++r) {
            int col = (tid >> 2) + r * 64, ko = (tid & 3) * 8;
            async_g2l_b128(ldsAddr(&Bb[buf][col * 32 + ko]), WbT,
                           (unsigned)(((n0 + col) * 512 + k0 + ko) * 2));
        }
    };

    stage(0, 0);
    wait_async_0();
    __syncthreads();

    int buf = 0;
    for (int kt = 0; kt < 16; ++kt) {
        if (kt < 15) stage(buf ^ 1, (kt + 1) * 32);

        FragB16 afr[2], bfr[4];
        for (int mi = 0; mi < 2; ++mi) {
            const __bf16* ap = &Ab[buf][(wm + mi * 16 + l15) * 32 + 8 * hl];
            afr[mi].q2[0] = *(const int4*)(ap);
            afr[mi].q2[1] = *(const int4*)(ap + 16);
        }
        for (int ni = 0; ni < 4; ++ni) {
            const __bf16* bp = &Bb[buf][(wn + ni * 16 + l15) * 32 + 8 * hl];
            bfr[ni].q2[0] = *(const int4*)(bp);
            bfr[ni].q2[1] = *(const int4*)(bp + 16);
        }
        for (int mi = 0; mi < 2; ++mi)
            for (int ni = 0; ni < 4; ++ni)
                acc[mi][ni] = __builtin_amdgcn_wmma_f32_16x16x32_bf16(
                    false, afr[mi].v, false, bfr[ni].v,
                    (short)0, acc[mi][ni], false, false);

        wait_async_0();
        __syncthreads();
        buf ^= 1;
    }

    // store C + bias  (C layout: vgpr r -> M = r + 8*hl ; N = l15)
    for (int mi = 0; mi < 2; ++mi)
        for (int ni = 0; ni < 4; ++ni) {
            int n  = n0 + wn + ni * 16 + l15;
            float bv = bias[n];
            for (int r = 0; r < 8; ++r) {
                int m = m0 + wm + mi * 16 + r + 8 * hl;
                XW[(size_t)m * G4 + n] = acc[mi][ni][r] + bv;
            }
        }
}

// ---------------------------------------------------------------------------
// K2: persistent recurrence. 16 WGs x 256 threads. WG g owns hidden cols
//     [wg*32, wg*32+32): gate cols {gate*512 + wg*32 + j}.
// ---------------------------------------------------------------------------
__global__ __launch_bounds__(256) void lstm_rec_kernel(
    const float* __restrict__ XW, const __bf16* __restrict__ UbT,
    __bf16* __restrict__ hbuf, unsigned* __restrict__ ctr,
    float* __restrict__ out) {

    __shared__ __bf16 Ut[128 * 512];   // 128KB: resident U slice (transposed, K contig)
    __shared__ __bf16 hs[32 * 512];    // 32KB : h_t
    __shared__ float  xws[32 * 128];   // 16KB : xW chunk for this step
    __shared__ float  gt[32 * 128];    // 16KB : h@U result
    __shared__ float  cs[32 * 32];     // 4KB  : cell state slice

    const int tid  = threadIdx.x;
    const int wg   = blockIdx.x;       // 0..15
    const int wave = tid >> 5;
    const int lane = tid & 31;
    const int hl   = lane >> 4;
    const int l15  = lane & 15;

    // Stage resident U slice: local col c = gate*32 + j -> global col gate*512 + wg*32 + j
    for (int i = tid; i < 128 * 32; i += 256) {       // 4096 x 16B chunks
        int c  = i >> 5;
        int ko = (i & 31) * 8;
        int g  = c >> 5, jc = c & 31;
        int gcol = g * 512 + wg * 32 + jc;
        async_g2l_b128(ldsAddr(&Ut[c * 512 + ko]), UbT,
                       (unsigned)((gcol * 512 + ko) * 2));
    }
    for (int i = tid; i < 32 * 32; i += 256)  cs[i] = 0.0f;
    for (int i = tid; i < 32 * 512; i += 256) hs[i] = (__bf16)0.0f;  // h_0 = 0
    wait_async_0();
    __syncthreads();

    for (int t = 0; t < T_STEPS; ++t) {
        // ---- async stage: h_t first (8 instrs), then xW chunk (4 instrs) ----
        if (t > 0) {
            const __bf16* hsrc = hbuf + (size_t)(t & 1) * (32 * 512);
            for (int i = tid; i < 2048; i += 256) {          // 2048 x 16B
                int off = i * 8;
                async_g2l_b128(ldsAddr(&hs[off]), hsrc, (unsigned)(off * 2));
            }
        }
        for (int i = tid; i < 1024; i += 256) {              // 1024 x 16B (32x128 f32)
            int row = i >> 5;
            int c   = (i & 31) * 4;
            int g   = c >> 5, jc = c & 31;
            size_t goff = ((size_t)t * B_SZ + row) * G4 + g * 512 + wg * 32 + jc;
            async_g2l_b128(ldsAddr(&xws[row * 128 + c]), XW, (unsigned)(goff * 4));
        }

        wait_async_le4();        // h done (in-order); xW may still be in flight
        __syncthreads();

        // ---- h(32x512) @ Uslice(512x128): wave owns 16-col group, mi = 0,1 ----
        v8f a0 = (v8f)0.0f, a1 = (v8f)0.0f;
        const int ccol = wave * 16 + l15;
        for (int k0 = 0; k0 < 512; k0 += 32) {
            FragB16 bfr;
            const __bf16* bp = &Ut[ccol * 512 + k0 + 8 * hl];
            bfr.q2[0] = *(const int4*)(bp);
            bfr.q2[1] = *(const int4*)(bp + 16);

            FragB16 af0, af1;
            const __bf16* ap0 = &hs[l15 * 512 + k0 + 8 * hl];
            af0.q2[0] = *(const int4*)(ap0);
            af0.q2[1] = *(const int4*)(ap0 + 16);
            const __bf16* ap1 = &hs[(16 + l15) * 512 + k0 + 8 * hl];
            af1.q2[0] = *(const int4*)(ap1);
            af1.q2[1] = *(const int4*)(ap1 + 16);

            a0 = __builtin_amdgcn_wmma_f32_16x16x32_bf16(
                false, af0.v, false, bfr.v, (short)0, a0, false, false);
            a1 = __builtin_amdgcn_wmma_f32_16x16x32_bf16(
                false, af1.v, false, bfr.v, (short)0, a1, false, false);
        }
        for (int r = 0; r < 8; ++r) {
            int col = wave * 16 + l15;
            gt[(r + 8 * hl) * 128 + col]      = a0[r];
            gt[(16 + r + 8 * hl) * 128 + col] = a1[r];
        }

        wait_async_0();          // xW chunk landed (overlapped with WMMA loop)
        __syncthreads();

        // ---- gates -> c,h (32x32 slice, 4 elems/thread) ----
        for (int i = tid; i < 32 * 32; i += 256) {
            int row = i >> 5, j = i & 31;
            float gi = gt[row * 128 + j]      + xws[row * 128 + j];
            float gf = gt[row * 128 + 32 + j] + xws[row * 128 + 32 + j];
            float gg = gt[row * 128 + 64 + j] + xws[row * 128 + 64 + j];
            float go = gt[row * 128 + 96 + j] + xws[row * 128 + 96 + j];
            float iv = sigmoidf_(gi);
            float fv = sigmoidf_(gf);
            float gv = tanhf(gg);
            float ov = sigmoidf_(go);
            float c  = fv * cs[i] + iv * gv;
            cs[i] = c;
            float h = ov * tanhf(c);
            int gcol = wg * 32 + j;
            out[((size_t)t * B_SZ + row) * H_SZ + gcol] = h;
            hbuf[(size_t)((t + 1) & 1) * (32 * 512) + row * 512 + gcol] = (__bf16)h;
            if (t == T_STEPS - 1) {
                out[(size_t)T_STEPS * B_SZ * H_SZ + row * H_SZ + gcol] = h;                       // h_T
                out[(size_t)T_STEPS * B_SZ * H_SZ + B_SZ * H_SZ + row * H_SZ + gcol] = c;         // c_T
            }
        }
        __syncthreads();
        __threadfence();

        // ---- grid barrier across the 16 persistent WGs ----
        if (tid == 0) {
            __hip_atomic_fetch_add(ctr, 1u, __ATOMIC_RELEASE, __HIP_MEMORY_SCOPE_AGENT);
            unsigned target = 16u * (unsigned)(t + 1);
            while (__hip_atomic_load(ctr, __ATOMIC_ACQUIRE, __HIP_MEMORY_SCOPE_AGENT) < target) {
                __builtin_amdgcn_s_sleep(1);
            }
        }
        __syncthreads();
    }
}

// ---------------------------------------------------------------------------
extern "C" void kernel_launch(void* const* d_in, const int* in_sizes, int n_in,
                              void* d_out, int out_size, void* d_ws, size_t ws_size,
                              hipStream_t stream) {
    const float* x    = (const float*)d_in[0];
    const float* W    = (const float*)d_in[1];
    const float* U    = (const float*)d_in[2];
    const float* bias = (const float*)d_in[3];
    float* out = (float*)d_out;

    char* ws = (char*)d_ws;
    float*  XW   = (float*)(ws + XW_OFF);
    __bf16* Xbf  = (__bf16*)(ws + XBF_OFF);
    __bf16* WbT  = (__bf16*)(ws + WBT_OFF);
    __bf16* UbT  = (__bf16*)(ws + UBT_OFF);
    __bf16* hbuf = (__bf16*)(ws + HBUF_OFF);
    unsigned* ctr = (unsigned*)(ws + CTR_OFF);

    hipMemsetAsync(hbuf, 0, 2 * 32 * 512 * sizeof(__bf16), stream);  // h_0 = 0
    hipMemsetAsync(ctr, 0, 64, stream);                              // barrier counter

    cvt_x_kernel<<<16384, 256, 0, stream>>>(x, Xbf, T_STEPS * B_SZ * H_SZ);
    cvt_tr_kernel<<<4096, 256, 0, stream>>>(W, WbT);
    cvt_tr_kernel<<<4096, 256, 0, stream>>>(U, UbT);

    dim3 g1(G4 / 256, (T_STEPS * B_SZ) / 64);
    xw_gemm_kernel<<<g1, 256, 0, stream>>>(Xbf, WbT, bias, XW);

    lstm_rec_kernel<<<16, 256, 0, stream>>>(XW, UbT, hbuf, ctr, out);
}
